// AttentionLayer_52286931861614
// MI455X (gfx1250) — matrix operand order repaired and checked by reading, not compiled, not run on hardware
//
#include <hip/hip_runtime.h>

// ---------------------------------------------------------------------------
// T5 attention layer for MI455X (gfx1250), all matmuls via v_wmma_f32_16x16x32_bf16
// B=4, S=2048, D=1024, H=16, DK=64, NUM_BUCKETS=32, MAX_DISTANCE=128
// GEMMs stage weight tiles in LDS with global_load_async_to_lds_b128 (ASYNCcnt),
// double-buffered so WMMAs never drain global loads.
// ---------------------------------------------------------------------------

typedef __attribute__((ext_vector_type(16))) __bf16 v16bf;
typedef __attribute__((ext_vector_type(8)))  __bf16 v8bf;
typedef __attribute__((ext_vector_type(8)))  float  v8f;
typedef __attribute__((ext_vector_type(4)))  float  v4f;

constexpr int Bn   = 4;
constexpr int Sn   = 2048;
constexpr int Dn   = 1024;
constexpr int Hn   = 16;
constexpr int DKn  = 64;
constexpr int MR   = Bn * Sn;       // 8192 rows for projection GEMMs
constexpr int SROW = 2052;          // LDS score row stride in floats (bank-skewed)
constexpr int BSTR = 40;            // LDS B-tile row stride in bf16 (80 B, bank-skewed)

// ---------------------------------------------------------------------------
// Prep kernels
// ---------------------------------------------------------------------------
__global__ void cvt_f32_bf16(const float* __restrict__ in, __bf16* __restrict__ out, int n) {
    int i = blockIdx.x * blockDim.x + threadIdx.x;
    int stride = gridDim.x * blockDim.x;
    for (; i < n; i += stride) out[i] = (__bf16)in[i];
}

// Wt[n][k] = W[k][n], fp32 -> bf16, 32x32 LDS tile transpose, 256 threads
__global__ void transpose_to_bf16(const float* __restrict__ W, __bf16* __restrict__ Wt) {
    __shared__ float tile[32][33];
    const int k0 = blockIdx.x * 32;
    const int n0 = blockIdx.y * 32;
    const int tx = threadIdx.x & 31;
    const int ty = threadIdx.x >> 5;           // 0..7
    for (int i = ty; i < 32; i += 8)
        tile[i][tx] = W[(size_t)(k0 + i) * Dn + n0 + tx];
    __syncthreads();
    for (int i = ty; i < 32; i += 8)
        Wt[(size_t)(n0 + i) * Dn + k0 + tx] = (__bf16)tile[tx][i];
}

// biasRel[h][p], p = (k - q) + 2047 in [0, 4094]; T5 bidirectional bucketing.
__global__ void build_bias(const float* __restrict__ table, float* __restrict__ biasRel) {
    int idx = blockIdx.x * blockDim.x + threadIdx.x;   // h*4096 + p
    if (idx >= Hn * 4096) return;
    int h = idx >> 12, p = idx & 4095;
    float v = 0.0f;
    if (p < 4095) {
        int rel = p - 2047;                    // k - q
        int bucket = (rel > 0) ? 16 : 0;       // num_buckets//2 = 16
        int rp = rel < 0 ? -rel : rel;
        if (rp < 8) {                          // max_exact = 8
            bucket += rp;
        } else {
            int lg = 8 + (int)(__logf((float)rp * 0.125f) * (8.0f / __logf(16.0f)));
            bucket += (lg < 15 ? lg : 15);
        }
        v = table[bucket * Hn + h];
    }
    biasRel[idx] = v;
}

// ---------------------------------------------------------------------------
// WMMA GEMM:  C[8192 x 1024] = A[8192 x 1024] * B[1024 x 1024]
//   A bf16 row-major, Bt bf16 = B^T row-major ([N][K]).
//   mode 0: store bf16 head-major        [(b*H+h)][s][d]
//   mode 2: store bf16 head-major transp [(b*H+h)][d][s]
//   mode 3: store fp32 plain             [row][col]
// Workgroup: 256 thr = 8 waves, 128x128 tile, wave -> 16 rows x 128 cols.
// B tile (128x32 bf16) staged in LDS by async-to-LDS DMA, double buffered.
// ---------------------------------------------------------------------------
__global__ __launch_bounds__(256)
void gemm_bf16_wmma(const __bf16* __restrict__ A, const __bf16* __restrict__ Bt,
                    void* __restrict__ Cout, int mode)
{
    __shared__ __bf16 btile[2][128 * BSTR];    // 2 x 10 KB, row stride 80 B

    const int tid  = threadIdx.x;
    const int wave = tid >> 5;
    const int lane = tid & 31;
    const int half = lane >> 4;
    const int lrow = lane & 15;

    const int m0 = blockIdx.x * 128 + wave * 16;
    const int n0 = blockIdx.y * 128;

    v8f acc[8] = {};
    const __bf16* aRow = A + (size_t)(m0 + lrow) * Dn;

    // Async-stage one 128x32 B tile: 512 16B segments, 2 per thread.
    auto stage = [&](int buf, int k0) {
        #pragma unroll
        for (int s = 0; s < 2; ++s) {
            const int seg  = tid + s * 256;
            const int nrow = seg >> 2;         // 0..127
            const int part = seg & 3;          // 16B segment within 64B row
            const __bf16* g = Bt + (size_t)(n0 + nrow) * Dn + k0 + part * 8;
            const unsigned lofs = (unsigned)(size_t)&btile[buf][nrow * BSTR + part * 8];
            asm volatile("global_load_async_to_lds_b128 %0, %1, off"
                         :: "v"(lofs), "v"(g) : "memory");
        }
    };

    stage(0, 0);
    asm volatile("s_wait_asynccnt 0x0" ::: "memory");
    __syncthreads();

    for (int kc = 0; kc < 32; ++kc) {
        const int k0 = kc * 32;
        const int p  = kc & 1;
        if (kc < 31) stage(p ^ 1, k0 + 32);    // prefetch next tile into other buffer

        __builtin_prefetch(aRow + k0 + 128, 0, 0);   // global_prefetch_b8 (A stream)
        // A fragment: row = m0+lrow, K = k0+half*16 .. +15 (contiguous)
        v8bf a_lo = *(const v8bf*)(aRow + k0 + half * 16);
        v8bf a_hi = *(const v8bf*)(aRow + k0 + half * 16 + 8);
        v16bf af;
        #pragma unroll
        for (int i = 0; i < 8; ++i) { af[i] = a_lo[i]; af[8 + i] = a_hi[i]; }

        #pragma unroll
        for (int nt = 0; nt < 8; ++nt) {
            // B fragment from LDS: n = nt*16+lrow, K = half*16 .. +15 (contiguous)
            const __bf16* bptr = &btile[p][(nt * 16 + lrow) * BSTR + half * 16];
            v8bf b_lo = *(const v8bf*)(bptr);
            v8bf b_hi = *(const v8bf*)(bptr + 8);
            v16bf bf;
            #pragma unroll
            for (int i = 0; i < 8; ++i) { bf[i] = b_lo[i]; bf[8 + i] = b_hi[i]; }
            acc[nt] = __builtin_amdgcn_wmma_f32_16x16x32_bf16(
                false, af, false, bf, (short)0, acc[nt], false, false);
        }

        asm volatile("s_wait_asynccnt 0x0" ::: "memory");
        __syncthreads();
    }

    // C layout: VGPR r -> row m0 + r + 8*half, col n0 + nt*16 + lrow
    #pragma unroll
    for (int nt = 0; nt < 8; ++nt) {
        const int col = n0 + nt * 16 + lrow;
        const int hh  = col >> 6;          // head
        const int dd  = col & 63;          // dim in head
        #pragma unroll
        for (int r = 0; r < 8; ++r) {
            const int row = m0 + r + 8 * half;
            const float v = acc[nt][r];
            const int bb = row >> 11;      // batch (row / 2048)
            const int ss = row & 2047;
            if (mode == 0) {
                ((__bf16*)Cout)[((size_t)(bb * Hn + hh) * Sn + ss) * DKn + dd] = (__bf16)v;
            } else if (mode == 2) {
                ((__bf16*)Cout)[((size_t)(bb * Hn + hh) * DKn + dd) * Sn + ss] = (__bf16)v;
            } else {
                ((float*)Cout)[(size_t)row * Dn + col] = v;
            }
        }
    }
}

// ---------------------------------------------------------------------------
// Attention kernel: one workgroup (128 thr = 4 waves) per (b,h,q-tile of 16).
//   Phase 1: scores = Q*K^T + bias -> LDS  (each wave: 512 keys, WMMA)
//   Phase 2: row softmax over 2048 keys (cooperative, exp kept in LDS)
//   Phase 3: O = P*V via WMMA (wave w -> dims [16w,16w+16)), scaled by 1/rowsum
// LDS: 16 x SROW fp32 scores (~128 KB of the 320 KB WGP LDS) + reduce scratch.
// ---------------------------------------------------------------------------
__global__ __launch_bounds__(128)
void attention_wmma(const __bf16* __restrict__ Qh,    // [(b*H+h)][S][64]
                    const __bf16* __restrict__ Kh,    // [(b*H+h)][S][64]
                    const __bf16* __restrict__ Vt,    // [(b*H+h)][64][S]
                    const float*  __restrict__ biasRel, // [H][4096]
                    __bf16* __restrict__ AO)          // [b*S + q][H*DK]
{
    extern __shared__ float lds[];
    float* scores = lds;                    // [16][SROW]
    float* partmx = lds + 16 * SROW;        // [16][8]
    float* partsm = partmx + 128;           // [16][8]
    float* rowmx  = partsm + 128;           // [16]
    float* rinv   = rowmx + 16;             // [16]

    const int wave = threadIdx.x >> 5;      // 0..3
    const int lane = threadIdx.x & 31;
    const int half = lane >> 4;
    const int lrow = lane & 15;

    const int qt = blockIdx.x & 127;        // S/16 tiles
    const int bh = blockIdx.x >> 7;         // 0..63
    const int h  = bh & 15;
    const int bb = bh >> 4;
    const int q0 = qt * 16;

    const __bf16* Qbase = Qh + (size_t)bh * Sn * DKn;
    const __bf16* Kbase = Kh + (size_t)bh * Sn * DKn;
    const __bf16* Vbase = Vt + (size_t)bh * DKn * Sn;
    const float*  bias  = biasRel + h * 4096;

    // Q A-fragments for K-chunks 0 and 32 (held in VGPRs for the whole phase)
    v16bf qf[2];
    {
        const __bf16* qrow = Qbase + (size_t)(q0 + lrow) * DKn;
        #pragma unroll
        for (int c = 0; c < 2; ++c) {
            v8bf lo = *(const v8bf*)(qrow + c * 32 + half * 16);
            v8bf hi = *(const v8bf*)(qrow + c * 32 + half * 16 + 8);
            #pragma unroll
            for (int i = 0; i < 8; ++i) { qf[c][i] = lo[i]; qf[c][8 + i] = hi[i]; }
        }
    }

    // ---- Phase 1: scores -------------------------------------------------
    for (int kt = 0; kt < 32; ++kt) {
        const int key0 = wave * 512 + kt * 16;
        const __bf16* krow = Kbase + (size_t)(key0 + lrow) * DKn;  // B = K^T -> Bt = K
        __builtin_prefetch(krow + 16 * DKn, 0, 0);                 // next key tile

        // hoist both K-chunk fragments before the two WMMAs
        v8bf l0 = *(const v8bf*)(krow + half * 16);
        v8bf h0 = *(const v8bf*)(krow + half * 16 + 8);
        v8bf l1 = *(const v8bf*)(krow + 32 + half * 16);
        v8bf h1 = *(const v8bf*)(krow + 32 + half * 16 + 8);
        v16bf bf0, bf1;
        #pragma unroll
        for (int i = 0; i < 8; ++i) {
            bf0[i] = l0[i]; bf0[8 + i] = h0[i];
            bf1[i] = l1[i]; bf1[8 + i] = h1[i];
        }
        v8f c = {};
        c = __builtin_amdgcn_wmma_f32_16x16x32_bf16(false, qf[0], false, bf0, (short)0, c, false, false);
        c = __builtin_amdgcn_wmma_f32_16x16x32_bf16(false, qf[1], false, bf1, (short)0, c, false, false);

        const int kcol = key0 + lrow;
        #pragma unroll
        for (int r = 0; r < 8; ++r) {
            const int row = r + 8 * half;
            const int q   = q0 + row;
            scores[row * SROW + kcol] = c[r] + bias[kcol - q + 2047];
        }
    }
    __syncthreads();

    // ---- Phase 2: softmax ------------------------------------------------
    {
        const int row = threadIdx.x >> 3;   // 0..15
        const int sub = threadIdx.x & 7;
        float m = -3.0e38f;
        for (int cidx = sub; cidx < Sn; cidx += 8)
            m = fmaxf(m, scores[row * SROW + cidx]);
        partmx[row * 8 + sub] = m;
        __syncthreads();
        if (threadIdx.x < 16) {
            float mm = partmx[threadIdx.x * 8];
            for (int i = 1; i < 8; ++i) mm = fmaxf(mm, partmx[threadIdx.x * 8 + i]);
            rowmx[threadIdx.x] = mm;
        }
        __syncthreads();
        const float rm = rowmx[row];
        float s = 0.0f;
        for (int cidx = sub; cidx < Sn; cidx += 8) {
            float e = __expf(scores[row * SROW + cidx] - rm);
            scores[row * SROW + cidx] = e;
            s += e;
        }
        partsm[row * 8 + sub] = s;
        __syncthreads();
        if (threadIdx.x < 16) {
            float ss = 0.0f;
            for (int i = 0; i < 8; ++i) ss += partsm[threadIdx.x * 8 + i];
            rinv[threadIdx.x] = 1.0f / ss;
        }
        __syncthreads();
    }

    // ---- Phase 3: O = P * V ---------------------------------------------
    {
        v8f acc = {};
        const int d0 = wave * 16;
        const __bf16* vrow = Vbase + (size_t)(d0 + lrow) * Sn;   // Bt = V^T rows
        for (int k0 = 0; k0 < Sn; k0 += 32) {
            // B fragment (V) and A fragment (P from LDS) hoisted together
            v8bf lo = *(const v8bf*)(vrow + k0 + half * 8);
            v8bf hi = *(const v8bf*)(vrow + k0 + 16 + half * 8);
            const float* prow = scores + lrow * SROW + k0 + half * 8;
            v4f p0 = *(const v4f*)(prow);
            v4f p1 = *(const v4f*)(prow + 4);
            v4f p2 = *(const v4f*)(prow + 16);
            v4f p3 = *(const v4f*)(prow + 20);
            v16bf af;
            #pragma unroll
            for (int i = 0; i < 4; ++i) {
                af[i]      = (__bf16)p0[i];
                af[4 + i]  = (__bf16)p1[i];
                af[8 + i]  = (__bf16)p2[i];
                af[12 + i] = (__bf16)p3[i];
            }
            v16bf bf;
            #pragma unroll
            for (int i = 0; i < 8; ++i) { bf[i] = lo[i]; bf[8 + i] = hi[i]; }
            acc = __builtin_amdgcn_wmma_f32_16x16x32_bf16(
                false, af, false, bf, (short)0, acc, false, false);
        }
        const int col = h * DKn + d0 + lrow;
        #pragma unroll
        for (int r = 0; r < 8; ++r) {
            const int row = r + 8 * half;
            const float v = acc[r] * rinv[row];
            AO[(size_t)(bb * Sn + q0 + row) * Dn + col] = (__bf16)v;
        }
    }
}

// ---------------------------------------------------------------------------
// Host-side launch
// ---------------------------------------------------------------------------
extern "C" void kernel_launch(void* const* d_in, const int* in_sizes, int n_in,
                              void* d_out, int out_size, void* d_ws, size_t ws_size,
                              hipStream_t stream) {
    (void)in_sizes; (void)n_in; (void)out_size; (void)ws_size;
    const float* hs  = (const float*)d_in[0];
    const float* Wq  = (const float*)d_in[1];
    const float* Wk  = (const float*)d_in[2];
    const float* Wv  = (const float*)d_in[3];
    const float* Wo  = (const float*)d_in[4];
    const float* tbl = (const float*)d_in[5];

    char* ws = (char*)d_ws;
    size_t off = 0;
    auto alloc = [&](size_t bytes) -> void* {
        off = (off + 255) & ~(size_t)255;
        void* p = ws + off;
        off += bytes;
        return p;
    };
    __bf16* Xbf = (__bf16*)alloc((size_t)MR * Dn * 2);       // 16 MB
    __bf16* Wqt = (__bf16*)alloc((size_t)Dn * Dn * 2);       //  2 MB each
    __bf16* Wkt = (__bf16*)alloc((size_t)Dn * Dn * 2);
    __bf16* Wvt = (__bf16*)alloc((size_t)Dn * Dn * 2);
    __bf16* Wot = (__bf16*)alloc((size_t)Dn * Dn * 2);
    __bf16* Qh  = (__bf16*)alloc((size_t)MR * Dn * 2);       // 16 MB
    __bf16* Kh  = (__bf16*)alloc((size_t)MR * Dn * 2);
    __bf16* Vtb = (__bf16*)alloc((size_t)MR * Dn * 2);
    __bf16* AO  = (__bf16*)alloc((size_t)MR * Dn * 2);
    float*  biasRel = (float*)alloc((size_t)Hn * 4096 * 4);  // 256 KB

    // prep
    cvt_f32_bf16<<<2048, 256, 0, stream>>>(hs, Xbf, MR * Dn);
    dim3 tg(Dn / 32, Dn / 32);
    transpose_to_bf16<<<tg, 256, 0, stream>>>(Wq, Wqt);
    transpose_to_bf16<<<tg, 256, 0, stream>>>(Wk, Wkt);
    transpose_to_bf16<<<tg, 256, 0, stream>>>(Wv, Wvt);
    transpose_to_bf16<<<tg, 256, 0, stream>>>(Wo, Wot);
    build_bias<<<(Hn * 4096) / 256, 256, 0, stream>>>(tbl, biasRel);

    // projections
    dim3 gg(MR / 128, Dn / 128);
    gemm_bf16_wmma<<<gg, 256, 0, stream>>>(Xbf, Wqt, Qh, 0);
    gemm_bf16_wmma<<<gg, 256, 0, stream>>>(Xbf, Wkt, Kh, 0);
    gemm_bf16_wmma<<<gg, 256, 0, stream>>>(Xbf, Wvt, Vtb, 2);

    // attention
    const size_t ldsBytes = (size_t)(16 * SROW + 128 + 128 + 16 + 16) * sizeof(float);
    attention_wmma<<<Bn * Hn * (Sn / 16), 128, ldsBytes, stream>>>(Qh, Kh, Vtb, biasRel, AO);

    // output projection (fp32 result)
    gemm_bf16_wmma<<<gg, 256, 0, stream>>>(AO, Wot, d_out, 3);
}
